// MinGRU_60421599920446
// MI455X (gfx1250) — matrix operand order repaired and checked by reading, not compiled, not run on hardware
//
#include <hip/hip_runtime.h>

typedef __attribute__((ext_vector_type(16))) _Float16 v16h;
typedef __attribute__((ext_vector_type(8)))  _Float16 v8h;
typedef __attribute__((ext_vector_type(8)))  float    v8f;
typedef __attribute__((ext_vector_type(4)))  float    v4f;

#define B_    8
#define S_    8192
#define D_    256
#define H_    256
#define CS    64          // s-chunk length
#define HB    32          // h columns per workgroup
#define XSTR  264         // padded f16 row stride for x chunk (256 + 8)
#define WSTR  264         // padded f16 row stride for weights
#define SBSTR 34          // padded f32 row stride for scan buffers

__global__ __launch_bounds__(256)
void mingru_fused(const float* __restrict__ x,
                  const float* __restrict__ h0,
                  const float* __restrict__ whw,
                  const float* __restrict__ whb,
                  const float* __restrict__ wzw,
                  const float* __restrict__ wzb,
                  float* __restrict__ out)
{
    // f32 staging for async global->LDS copies of the x chunk (64 KB)
    __shared__ __attribute__((aligned(16))) float    sStg[CS * D_];
    __shared__ __attribute__((aligned(16))) _Float16 sWz [HB * WSTR];
    __shared__ __attribute__((aligned(16))) _Float16 sWh [HB * WSTR];
    __shared__ __attribute__((aligned(16))) _Float16 sX  [CS * XSTR];
    __shared__ __attribute__((aligned(16))) float    sZ  [CS * SBSTR];
    __shared__ __attribute__((aligned(16))) float    sHt [CS * SBSTR];

    const int tid   = threadIdx.x;
    const int lane  = tid & 31;
    const int wave  = tid >> 5;
    const int hb    = blockIdx.x;        // 0..7  (H/HB)
    const int b     = blockIdx.y;        // 0..7  (B)
    const int hbase = hb * HB;

    // low 32 bits of a generic pointer into the shared aperture == LDS offset
    const uint32_t stgBase = (uint32_t)(uintptr_t)(void*)sStg;

    // ---- kick off async copy of chunk 0: 4096 float4, 16 per thread ----
    {
        const float* gsrc = x + (size_t)b * S_ * D_;
        #pragma unroll
        for (int i = 0; i < 16; ++i) {
            int f4 = i * 256 + tid;                       // lanes contiguous: 512B/wave
            const float* ga = gsrc + (size_t)f4 * 4;
            uint32_t la = stgBase + (uint32_t)f4 * 16;
            asm volatile("global_load_async_to_lds_b128 %0, %1, off"
                         :: "v"(la), "v"(ga) : "memory");
        }
    }

    // ---- load + convert weights to f16 LDS (32 rows x 256 cols each) ----
    #pragma unroll
    for (int i = 0; i < 8; ++i) {
        int f4  = i * 256 + tid;         // float4 index, 0..2047
        int row = f4 >> 6;               // 64 float4 per row
        int c4  = f4 & 63;
        v4f gz = *(const v4f*)(wzw + (size_t)(hbase + row) * D_ + c4 * 4);
        v4f gh = *(const v4f*)(whw + (size_t)(hbase + row) * D_ + c4 * 4);
        _Float16* pz = &sWz[row * WSTR + c4 * 4];
        _Float16* ph = &sWh[row * WSTR + c4 * 4];
        #pragma unroll
        for (int q = 0; q < 4; ++q) { pz[q] = (_Float16)gz[q]; ph[q] = (_Float16)gh[q]; }
    }

    // ---- per-wave tile jobs: job = wave (z matrix) and wave+8 (h_tilde) ----
    float biasv[2];
    #pragma unroll
    for (int jn = 0; jn < 2; ++jn) {
        int job = wave + jn * 8;
        int mat = job >> 3;
        int ht  = job & 1;
        int col = hbase + ht * 16 + (lane & 15);
        biasv[jn] = mat ? whb[col] : wzb[col];
    }

    // scan carry (meaningful only for tid < 32; lane = h column within block)
    float hcarry = 0.0f;
    if (tid < 32) hcarry = h0[(size_t)b * H_ + hbase + lane];

    for (int s0 = 0; s0 < S_; s0 += CS) {
        // ---- chunk s0's staged f32 data has landed in LDS ----
        asm volatile("s_wait_asynccnt 0x0" ::: "memory");
        __syncthreads();

        // ---- convert staging f32 -> padded f16 operand buffer ----
        #pragma unroll 4
        for (int i = 0; i < 16; ++i) {
            int f4  = i * 256 + tid;     // 4096 float4 = 64 rows * 64 f4
            int row = f4 >> 6;
            int c4  = f4 & 63;
            v4f g = *(const v4f*)(&sStg[(size_t)f4 * 4]);
            _Float16* p = &sX[row * XSTR + c4 * 4];
            #pragma unroll
            for (int q = 0; q < 4; ++q) p[q] = (_Float16)g[q];
        }
        __syncthreads();                 // staging free; sX ready

        // ---- issue async copy of NEXT chunk; overlaps WMMA + scan below ----
        if (s0 + CS < S_) {
            const float* gsrc = x + ((size_t)b * S_ + s0 + CS) * D_;
            #pragma unroll
            for (int i = 0; i < 16; ++i) {
                int f4 = i * 256 + tid;
                const float* ga = gsrc + (size_t)f4 * 4;
                uint32_t la = stgBase + (uint32_t)f4 * 16;
                asm volatile("global_load_async_to_lds_b128 %0, %1, off"
                             :: "v"(la), "v"(ga) : "memory");
            }
        }

        // ---- WMMA compute: 2 tile jobs per wave ----
        #pragma unroll
        for (int jn = 0; jn < 2; ++jn) {
            int job = wave + jn * 8;
            int mat = job >> 3;          // 0 = z, 1 = h_tilde
            int st  = (job & 7) >> 1;    // s-tile 0..3
            int ht  = job & 1;           // h-tile 0..1
            const _Float16* W = mat ? sWh : sWz;

            // A operand: lanes 0-15 hold rows M=0..15 with K {0..7,16..23};
            //            lanes 16-31 hold same rows with K {8..15,24..31}
            const int arow = st * 16 + (lane & 15);
            const int ak   = (lane >> 4) ? 8 : 0;
            // B operand: lanes 0-15: col N=lane, K {0..7,16..23}; lanes 16-31: K {8..15,24..31}
            const int bcol = ht * 16 + (lane & 15);
            const int bk   = (lane >> 4) ? 8 : 0;

            v8f acc = {};
            #pragma unroll
            for (int kb = 0; kb < 8; ++kb) {
                const int k0 = kb * 32;
                v8h alo = *(const v8h*)(&sX[arow * XSTR + k0 + ak]);
                v8h ahi = *(const v8h*)(&sX[arow * XSTR + k0 + ak + 16]);
                v16h a;
                #pragma unroll
                for (int q = 0; q < 8; ++q) { a[q] = alo[q]; a[q + 8] = ahi[q]; }
                v8h blo = *(const v8h*)(&W[bcol * WSTR + k0 + bk]);
                v8h bhi = *(const v8h*)(&W[bcol * WSTR + k0 + bk + 16]);
                v16h bm;
                #pragma unroll
                for (int q = 0; q < 8; ++q) { bm[q] = blo[q]; bm[q + 8] = bhi[q]; }
                acc = __builtin_amdgcn_wmma_f32_16x16x32_f16(
                    false, a, false, bm, (short)0, acc, false, false);
            }

            // C layout: VGPR r -> row M = r + 8*(lane>=16), col N = lane&15
            float* dst = mat ? sHt : sZ;
            const int ccol  = ht * 16 + (lane & 15);
            const int crow0 = st * 16 + ((lane >> 4) ? 8 : 0);
            #pragma unroll
            for (int r = 0; r < 8; ++r) {
                float v = acc[r] + biasv[jn];
                if (mat == 0) {
                    // fast sigmoid: v_exp_f32 + v_rcp_f32 (f16 GEMM dominates error)
                    v = __builtin_amdgcn_rcpf(1.0f + __expf(-v));
                }
                dst[(crow0 + r) * SBSTR + ccol] = v;
            }
        }
        __syncthreads();

        // ---- sequential scan over the chunk (wave 0, lane = h column) ----
        if (tid < 32) {
            float h = hcarry;
            size_t obase = ((size_t)b * S_ + s0) * H_ + hbase + lane;
            for (int s = 0; s < CS; ++s) {
                float z  = sZ [s * SBSTR + lane];
                float t  = sHt[s * SBSTR + lane];
                h = (1.0f - z) * h + z * t;
                out[obase + (size_t)s * H_] = h;
            }
            hcarry = h;
        }
        __syncthreads();   // scan buffers / sX reused next chunk
    }
}

extern "C" void kernel_launch(void* const* d_in, const int* in_sizes, int n_in,
                              void* d_out, int out_size, void* d_ws, size_t ws_size,
                              hipStream_t stream) {
    // setup_inputs order: x, h0, w_h_w, w_h_b, w_z_w, w_z_b  (all float32)
    const float* x   = (const float*)d_in[0];
    const float* h0  = (const float*)d_in[1];
    const float* whw = (const float*)d_in[2];
    const float* whb = (const float*)d_in[3];
    const float* wzw = (const float*)d_in[4];
    const float* wzb = (const float*)d_in[5];
    float* out = (float*)d_out;
    (void)in_sizes; (void)n_in; (void)out_size; (void)d_ws; (void)ws_size;

    dim3 grid(H_ / HB, B_);   // 8 x 8 = 64 workgroups
    mingru_fused<<<grid, 256, 0, stream>>>(x, h0, whw, whb, wzw, wzb, out);
}
